// Attention_44830868635854
// MI455X (gfx1250) — matrix operand order repaired and checked by reading, not compiled, not run on hardware
//
#include <hip/hip_runtime.h>
#include <hip/hip_bf16.h>

typedef __attribute__((ext_vector_type(16))) _Float16 v16h;
typedef __attribute__((ext_vector_type(8)))  _Float16 v8h;
typedef __attribute__((ext_vector_type(8)))  float    v8f;
typedef __attribute__((ext_vector_type(4)))  float    f4;

#define B_      4
#define DIM_    128
#define HW_     48
#define N_      2304      // 48*48
#define HEADS_  8
#define HD_     16
#define INNER_  128
#define GROUPS_ 8
#define EPS_    1e-5f
#define NT_     144       // N_/16 n-tiles

// ---- pack 16 f32 (from four float4) into a v16h A-operand ----
__device__ __forceinline__ v16h pack16(f4 f0, f4 f1, f4 f2, f4 f3) {
  v16h a;
  a[0]=(_Float16)f0[0];  a[1]=(_Float16)f0[1];  a[2]=(_Float16)f0[2];  a[3]=(_Float16)f0[3];
  a[4]=(_Float16)f1[0];  a[5]=(_Float16)f1[1];  a[6]=(_Float16)f1[2];  a[7]=(_Float16)f1[3];
  a[8]=(_Float16)f2[0];  a[9]=(_Float16)f2[1];  a[10]=(_Float16)f2[2]; a[11]=(_Float16)f2[3];
  a[12]=(_Float16)f3[0]; a[13]=(_Float16)f3[1]; a[14]=(_Float16)f3[2]; a[15]=(_Float16)f3[3];
  return a;
}

// ============ 1) x[b,c,n] (f32) -> xT[b,n,c] (f16) ============
__global__ __launch_bounds__(256) void k_xt(const float* __restrict__ x,
                                            _Float16* __restrict__ xT) {
  int idx = blockIdx.x * 256 + threadIdx.x;          // B*DIM*N threads exactly
  int n = idx % N_;
  int c = (idx / N_) & (DIM_ - 1);
  int b = idx / (N_ * DIM_);
  xT[(size_t)(b * N_ + n) * DIM_ + c] = (_Float16)x[idx];
}

// ============ 2) QKV GEMM: one wave per 16x16 output tile ============
__global__ __launch_bounds__(256) void k_qkv(const _Float16* __restrict__ xT,
                                             const float* __restrict__ w_qkv,
                                             const float* __restrict__ b_qkv,
                                             float* __restrict__ q_raw,
                                             _Float16* __restrict__ kT,
                                             _Float16* __restrict__ vm) {
  const int lane = threadIdx.x & 31;
  const int wid  = threadIdx.x >> 5;
  const int tile = blockIdx.x * 8 + wid;             // 4*24*144 = 13824 tiles
  const int b  = tile / (24 * NT_);
  const int r0 = tile % (24 * NT_);
  const int o_base = (r0 / NT_) * 16;
  const int n_base = (r0 % NT_) * 16;
  const int hi  = (lane >= 16);
  const int l16 = lane & 15;
  const int o_row = o_base + l16;
  const int n_col = n_base + l16;

  v8f acc = {};
  #pragma unroll
  for (int cb = 0; cb < DIM_; cb += 32) {
    // A: w_qkv 16(o) x 32(c), f32 -> f16, contiguous k = 16*hi + s (same as B)
    const float* wp = w_qkv + (size_t)o_row * DIM_ + cb + (hi ? 16 : 0);
    f4 f0 = *(const f4*)(wp);      f4 f1 = *(const f4*)(wp + 4);
    f4 f2 = *(const f4*)(wp + 8); f4 f3 = *(const f4*)(wp + 12);
    v16h a = pack16(f0, f1, f2, f3);
    // B: xT 32(c) x 16(n): contiguous 32B per lane
    v16h bm = *(const v16h*)(xT + (size_t)(b * N_ + n_col) * DIM_ + cb + (hi ? 16 : 0));
    acc = __builtin_amdgcn_wmma_f32_16x16x32_f16(false, a, false, bm,
                                                 (short)0, acc, false, false);
  }
  #pragma unroll
  for (int r = 0; r < 8; ++r) {
    int o = o_base + r + (hi ? 8 : 0);
    float v = acc[r] + b_qkv[o];
    if (o < 128) {                                   // q -> f32 for depthwise conv
      q_raw[(size_t)(b * DIM_ + o) * N_ + n_col] = v;
    } else if (o < 256) {                            // k -> kT[b,h,n,d] f16
      int c = o - 128, h = c >> 4, d = c & 15;
      kT[(size_t)((b * HEADS_ + h) * N_ + n_col) * HD_ + d] = (_Float16)v;
    } else {                                         // v -> v[b,h,d,n] f16
      int c = o - 256, h = c >> 4, d = c & 15;
      vm[(size_t)((b * HEADS_ + h) * HD_ + d) * N_ + n_col] = (_Float16)v;
    }
  }
}

// ============ 3) depthwise 3x3 + bias on q -> qT[b,h,n,d] f16 ============
__global__ __launch_bounds__(256) void k_dw(const float* __restrict__ q_raw,
                                            const float* __restrict__ w_dw,
                                            const float* __restrict__ b_dw,
                                            _Float16* __restrict__ qT) {
  int idx = blockIdx.x * 256 + threadIdx.x;          // B*128*N threads exactly
  int xc = idx % HW_;
  int yr = (idx / HW_) % HW_;
  int c  = (idx / N_) & (DIM_ - 1);
  int b  = idx / (N_ * DIM_);
  const float* base = q_raw + (size_t)(b * DIM_ + c) * N_;
  const float* w = w_dw + c * 9;
  float s = b_dw[c];
  #pragma unroll
  for (int ky = 0; ky < 3; ++ky) {
    int yy = yr + ky - 1;
    if (yy < 0 || yy >= HW_) continue;
    #pragma unroll
    for (int kx = 0; kx < 3; ++kx) {
      int xx = xc + kx - 1;
      if (xx < 0 || xx >= HW_) continue;
      s += base[yy * HW_ + xx] * w[ky * 3 + kx];
    }
  }
  int n = yr * HW_ + xc;
  int h = c >> 4, d = c & 15;
  qT[(size_t)((b * HEADS_ + h) * N_ + n) * HD_ + d] = (_Float16)s;
}

// ============ 4) flash attention, one wave per (b,h,16-row tile) ============
__global__ __launch_bounds__(256) void k_attn(const _Float16* __restrict__ qT,
                                              const _Float16* __restrict__ kT,
                                              const _Float16* __restrict__ vm,
                                              _Float16* __restrict__ aT) {
  __shared__ __align__(64) _Float16 lds[8][16 * 32];
  const int lane = threadIdx.x & 31;
  const int wid  = threadIdx.x >> 5;
  const int tile = blockIdx.x * 8 + wid;             // 4*8*144 = 4608 tiles
  const int b  = tile / (HEADS_ * NT_);
  const int h  = (tile / NT_) % HEADS_;
  const int n_base = (tile % NT_) * 16;
  const int hi  = (lane >= 16);
  const int l16 = lane & 15;
  const int bh  = b * HEADS_ + h;
  _Float16* P = lds[wid];

  // Q tile (A operand): rows n, K = d (16 real, 16 zero-pad)
  v16h aq = {};
  if (!hi) aq = *(const v16h*)(qT + (size_t)((size_t)bh * N_ + n_base + l16) * HD_);

  v8f oacc = {};
  float mrow[8], lrow[8];
  #pragma unroll
  for (int r = 0; r < 8; ++r) { mrow[r] = -1e30f; lrow[r] = 0.f; }

  const _Float16* kbase = kT + (size_t)bh * N_ * HD_;
  const _Float16* vbase = vm + (size_t)bh * HD_ * N_ + (size_t)l16 * N_;

  for (int mb = 0; mb < N_; mb += 32) {
    // K tiles (B operand): lanes 0-15 carry d=0..15, hi lanes are zero pad
    v16h bk0 = {}, bk1 = {};
    if (!hi) {
      bk0 = *(const v16h*)(kbase + (size_t)(mb + l16) * HD_);
      bk1 = *(const v16h*)(kbase + (size_t)(mb + 16 + l16) * HD_);
    }
    if (mb + 32 < N_) {
      __builtin_prefetch(kbase + (size_t)(mb + 32 + l16) * HD_, 0, 3);
      __builtin_prefetch(vbase + mb + 32 + (hi ? 16 : 0), 0, 3);
    }
    v8f cz = {};
    v8f s0 = __builtin_amdgcn_wmma_f32_16x16x32_f16(false, aq, false, bk0,
                                                    (short)0, cz, false, false);
    v8f s1 = __builtin_amdgcn_wmma_f32_16x16x32_f16(false, aq, false, bk1,
                                                    (short)0, cz, false, false);
    // V tile (B operand for PV): cols d = lane&15, K rows = m (contiguous)
    v16h bv = *(const v16h*)(vbase + mb + (hi ? 16 : 0));

    float p0[8], p1[8];
    #pragma unroll
    for (int r = 0; r < 8; ++r) {                    // row = r + 8*hi, col = lane&15
      float a0 = s0[r] * 0.25f, a1 = s1[r] * 0.25f;  // HEAD_DIM^-0.5
      float t = fmaxf(a0, a1);
      t = fmaxf(t, __shfl_xor(t, 1));
      t = fmaxf(t, __shfl_xor(t, 2));
      t = fmaxf(t, __shfl_xor(t, 4));
      t = fmaxf(t, __shfl_xor(t, 8));                // stays within 16-lane half
      float mnew  = fmaxf(mrow[r], t);
      float alpha = __expf(mrow[r] - mnew);
      mrow[r] = mnew;
      p0[r] = __expf(a0 - mnew);
      p1[r] = __expf(a1 - mnew);
      float rs = p0[r] + p1[r];
      rs += __shfl_xor(rs, 1);
      rs += __shfl_xor(rs, 2);
      rs += __shfl_xor(rs, 4);
      rs += __shfl_xor(rs, 8);
      lrow[r] = lrow[r] * alpha + rs;
      oacc[r] *= alpha;
    }
    // C-layout -> A-layout repack of P through LDS (intra-wave, DS is in-order)
    #pragma unroll
    for (int r = 0; r < 8; ++r) {
      int row = r + (hi ? 8 : 0);
      P[row * 32 + l16]      = (_Float16)p0[r];
      P[row * 32 + 16 + l16] = (_Float16)p1[r];
    }
    __builtin_amdgcn_wave_barrier();
    v8h plo = *(const v8h*)(P + l16 * 32 + (hi ? 16 : 0));
    v8h phi = *(const v8h*)(P + l16 * 32 + 8 + (hi ? 16 : 0));
    __builtin_amdgcn_wave_barrier();
    v16h ap = __builtin_shufflevector(plo, phi, 0,1,2,3,4,5,6,7,8,9,10,11,12,13,14,15);
    oacc = __builtin_amdgcn_wmma_f32_16x16x32_f16(false, ap, false, bv,
                                                  (short)0, oacc, false, false);
  }

  #pragma unroll
  for (int r = 0; r < 8; ++r) {                      // normalize + fused ReLU -> aT[b,n,c]
    float v = oacc[r] / lrow[r];
    v = fmaxf(v, 0.f);
    int n = n_base + r + (hi ? 8 : 0);
    aT[(size_t)(b * N_ + n) * INNER_ + h * HD_ + l16] = (_Float16)v;
  }
}

// ============ 5) zero GroupNorm stats ============
__global__ void k_zero(float* __restrict__ g) { g[threadIdx.x] = 0.f; }

// ============ 6) proj GEMM + per-(b,group) stats ============
__global__ __launch_bounds__(256) void k_proj(const _Float16* __restrict__ aT,
                                              const float* __restrict__ w_proj,
                                              float* __restrict__ outp,
                                              float* __restrict__ gsum,
                                              float* __restrict__ gssq) {
  const int lane = threadIdx.x & 31;
  const int wid  = threadIdx.x >> 5;
  const int tile = blockIdx.x * 8 + wid;             // 4*8*144 = 4608 tiles
  const int b  = tile / (GROUPS_ * NT_);
  const int g  = (tile / NT_) & 7;                   // group == 16-row o tile
  const int o_base = g * 16;
  const int n_base = (tile % NT_) * 16;
  const int hi  = (lane >= 16);
  const int l16 = lane & 15;
  const int o_row = o_base + l16;
  const int n_col = n_base + l16;

  v8f acc = {};
  #pragma unroll
  for (int cb = 0; cb < INNER_; cb += 32) {
    const float* wp = w_proj + (size_t)o_row * INNER_ + cb + (hi ? 16 : 0);
    f4 f0 = *(const f4*)(wp);      f4 f1 = *(const f4*)(wp + 4);
    f4 f2 = *(const f4*)(wp + 8); f4 f3 = *(const f4*)(wp + 12);
    v16h a = pack16(f0, f1, f2, f3);
    v16h bm = *(const v16h*)(aT + (size_t)(b * N_ + n_col) * INNER_ + cb + (hi ? 16 : 0));
    acc = __builtin_amdgcn_wmma_f32_16x16x32_f16(false, a, false, bm,
                                                 (short)0, acc, false, false);
  }
  float s = 0.f, ss = 0.f;
  #pragma unroll
  for (int r = 0; r < 8; ++r) {
    int o = o_base + r + (hi ? 8 : 0);
    float v = acc[r];
    outp[(size_t)(b * DIM_ + o) * N_ + n_col] = v;
    s += v; ss += v * v;
  }
  #pragma unroll
  for (int m = 1; m < 32; m <<= 1) { s += __shfl_xor(s, m); ss += __shfl_xor(ss, m); }
  if (lane == 0) {
    atomicAdd(&gsum[b * GROUPS_ + g], s);
    atomicAdd(&gssq[b * GROUPS_ + g], ss);
  }
}

// ============ 7) GroupNorm finalize ============
__global__ __launch_bounds__(256) void k_gn(const float* __restrict__ outp,
                                            const float* __restrict__ gsum,
                                            const float* __restrict__ gssq,
                                            const float* __restrict__ gn_w,
                                            const float* __restrict__ gn_b,
                                            float* __restrict__ out) {
  int idx = blockIdx.x * 256 + threadIdx.x;          // B*DIM*N threads exactly
  int c = (idx / N_) & (DIM_ - 1);
  int b = idx / (N_ * DIM_);
  int g = c >> 4;
  const float inv_cnt = 1.0f / (16.0f * N_);
  float mu  = gsum[b * GROUPS_ + g] * inv_cnt;
  float var = gssq[b * GROUPS_ + g] * inv_cnt - mu * mu;
  float x = outp[idx];
  out[idx] = (x - mu) * rsqrtf(var + EPS_) * gn_w[c] + gn_b[c];
}

extern "C" void kernel_launch(void* const* d_in, const int* in_sizes, int n_in,
                              void* d_out, int out_size, void* d_ws, size_t ws_size,
                              hipStream_t stream) {
  const float* x      = (const float*)d_in[0];
  const float* w_qkv  = (const float*)d_in[1];
  const float* b_qkv  = (const float*)d_in[2];
  const float* w_dw   = (const float*)d_in[3];
  const float* b_dw   = (const float*)d_in[4];
  const float* w_proj = (const float*)d_in[5];
  const float* gn_w   = (const float*)d_in[6];
  const float* gn_b   = (const float*)d_in[7];
  float* out = (float*)d_out;

  // workspace layout (bytes), all 4 KB aligned; total ~21.2 MB
  char* ws = (char*)d_ws;
  _Float16* xT   = (_Float16*)(ws + 0);          //  2,359,296 B
  float*    qraw = (float*)   (ws + 2359296);    //  4,718,592 B
  _Float16* qT   = (_Float16*)(ws + 7077888);    //  2,359,296 B
  _Float16* kT   = (_Float16*)(ws + 9437184);    //  2,359,296 B
  _Float16* vm   = (_Float16*)(ws + 11796480);   //  2,359,296 B
  _Float16* aT   = (_Float16*)(ws + 14155776);   //  2,359,296 B
  float*    outp = (float*)   (ws + 16515072);   //  4,718,592 B
  float*    gsum = (float*)   (ws + 21233664);   //  32 + 32 floats
  float*    gssq = gsum + 32;

  k_xt  <<<4608, 256, 0, stream>>>(x, xT);
  k_qkv <<<1728, 256, 0, stream>>>(xT, w_qkv, b_qkv, qraw, kT, vm);
  k_dw  <<<4608, 256, 0, stream>>>(qraw, w_dw, b_dw, qT);
  k_zero<<<1, 64, 0, stream>>>(gsum);
  k_attn<<<576, 256, 0, stream>>>(qT, kT, vm, aT);
  k_proj<<<576, 256, 0, stream>>>(aT, w_proj, outp, gsum, gssq);
  k_gn  <<<4608, 256, 0, stream>>>(outp, gsum, gssq, gn_w, gn_b, out);
}
